// GraphConvolution_25761213841714
// MI455X (gfx1250) — matrix operand order repaired and checked by reading, not compiled, not run on hardware
//
#include <hip/hip_runtime.h>

typedef __attribute__((ext_vector_type(16))) __bf16 v16bf;
typedef __attribute__((ext_vector_type(8)))  float  v8f;
typedef __attribute__((ext_vector_type(4)))  float  v4f;

#define IN_DIM  256
#define OUT_DIM 64

// -----------------------------------------------------------------------------
// Kernel 1: support = X (f32) * W (f32->bf16)  via v_wmma_f32_16x16x32_bf16
// 128 threads = 4 waves; each wave owns one 16-row tile and all 4 col-tiles.
// W is staged into LDS transposed [n][k] as bf16 (32KB) so B fragments are
// contiguous ds_load_b128 reads.
// -----------------------------------------------------------------------------
__global__ __launch_bounds__(128) void gcn_gemm_wmma(
    const float* __restrict__ input,
    const float* __restrict__ weight,
    float* __restrict__ support,
    int n_rows)
{
    __shared__ __bf16 ldsW[OUT_DIM * IN_DIM];   // [n][k], 32 KB

    const int tid = threadIdx.x;
    // Stage weight [k][n] (row-major) -> LDS [n][k] bf16
    for (int i = tid; i < IN_DIM * OUT_DIM; i += 128) {
        const int k = i >> 6;        // weight row (K)
        const int n = i & 63;        // weight col (N)
        ldsW[n * IN_DIM + k] = (__bf16)weight[i];
    }
    __syncthreads();

    const int wave    = tid >> 5;
    const int lane    = tid & 31;
    const int halfSel = lane >> 4;   // 0: lanes 0-15, 1: lanes 16-31
    const int mloc    = lane & 15;
    const int rowBlock = blockIdx.x * 4 + wave;
    if (rowBlock * 16 >= n_rows) return;   // wave-uniform exit (EXEC all-1s for WMMA)

    const float* arow = input + (size_t)(rowBlock * 16 + mloc) * IN_DIM;

    v8f acc[4] = {};

    #pragma unroll
    for (int ks = 0; ks < IN_DIM; ks += 32) {
        // ---- A fragment (16x32 bf16): lane holds K = ks + halfSel*16 + {0..15}
        const int kbase = ks + halfSel * 16;
        v4f a0 = *(const v4f*)(arow + kbase);        // K+0..3
        v4f a1 = *(const v4f*)(arow + kbase + 4);    // K+4..7
        v4f a2 = *(const v4f*)(arow + kbase + 8);    // K+8..11
        v4f a3 = *(const v4f*)(arow + kbase + 12);   // K+12..15
        v16bf afrag;
        #pragma unroll
        for (int j = 0; j < 4; ++j) {
            afrag[j]      = (__bf16)a0[j];
            afrag[4 + j]  = (__bf16)a1[j];
            afrag[8 + j]  = (__bf16)a2[j];
            afrag[12 + j] = (__bf16)a3[j];
        }

        // ---- 4 output col-tiles, B from LDS: K = ks + halfSel*16 + j (contiguous)
        #pragma unroll
        for (int t = 0; t < 4; ++t) {
            const int n = t * 16 + mloc;
            const v16bf bfrag =
                *(const v16bf*)(&ldsW[n * IN_DIM + ks + halfSel * 16]);
            acc[t] = __builtin_amdgcn_wmma_f32_16x16x32_bf16(
                /*neg_a=*/false, afrag, /*neg_b=*/false, bfrag,
                /*c_mod=*/(short)0, acc[t],
                /*reuse_a=*/false, /*reuse_b=*/false);
        }
    }

    // ---- store D: VGPR r -> row (r + halfSel*8), col = t*16 + (lane&15)
    #pragma unroll
    for (int t = 0; t < 4; ++t) {
        const int n = t * 16 + mloc;
        #pragma unroll
        for (int r = 0; r < 8; ++r) {
            const int m = r + halfSel * 8;
            support[(size_t)(rowBlock * 16 + m) * OUT_DIM + n] = acc[t][r];
        }
    }
}

// -----------------------------------------------------------------------------
// Kernel 2: out[i] = bias[i % 64]   (initializes poisoned d_out)
// -----------------------------------------------------------------------------
__global__ void gcn_bias_init(const float* __restrict__ bias,
                              float* __restrict__ out, int total)
{
    const int i = blockIdx.x * blockDim.x + threadIdx.x;
    if (i < total) out[i] = bias[i & 63];
}

// -----------------------------------------------------------------------------
// Kernel 3: scatter  out[dst] += support[src] * val
// 16 threads per edge, 4 cols (float4) per thread; hardware f32 atomic via asm
// so it cannot fall back to a CAS loop. support & out are L2-resident.
// -----------------------------------------------------------------------------
__device__ __forceinline__ void atomic_add_f32(float* p, float v) {
    asm volatile("global_atomic_add_f32 %0, %1, off" :: "v"(p), "v"(v) : "memory");
}

__global__ void gcn_scatter(const float* __restrict__ support,
                            const float* __restrict__ edge_val,
                            const int* __restrict__ edge_src,
                            const int* __restrict__ edge_dst,
                            float* __restrict__ out, int n_edges)
{
    const int gid = blockIdx.x * blockDim.x + threadIdx.x;
    const int e = gid >> 4;
    if (e >= n_edges) return;
    const int c   = (gid & 15) * 4;
    const int src = edge_src[e];
    const int dst = edge_dst[e];
    const float val = edge_val[e];

    const v4f s = *(const v4f*)(support + (size_t)src * OUT_DIM + c);
    float* o = out + (size_t)dst * OUT_DIM + c;
    atomic_add_f32(o + 0, s[0] * val);
    atomic_add_f32(o + 1, s[1] * val);
    atomic_add_f32(o + 2, s[2] * val);
    atomic_add_f32(o + 3, s[3] * val);
}

// -----------------------------------------------------------------------------
extern "C" void kernel_launch(void* const* d_in, const int* in_sizes, int n_in,
                              void* d_out, int out_size, void* d_ws, size_t ws_size,
                              hipStream_t stream)
{
    const float* input    = (const float*)d_in[0];
    const float* weight   = (const float*)d_in[1];
    const float* bias     = (const float*)d_in[2];
    const float* edge_val = (const float*)d_in[3];
    const int*   edge_src = (const int*)d_in[4];
    const int*   edge_dst = (const int*)d_in[5];
    float* out     = (float*)d_out;
    float* support = (float*)d_ws;     // n_rows * 64 f32 = 12.8 MB scratch

    const int n_rows  = in_sizes[0] / IN_DIM;   // 50000
    const int n_edges = in_sizes[3];            // 800000

    // 1) GEMM (WMMA bf16)
    const int rowBlocks = (n_rows + 15) / 16;       // 3125 (exact, no tail rows)
    const int gemmGrid  = (rowBlocks + 3) / 4;
    gcn_gemm_wmma<<<gemmGrid, 128, 0, stream>>>(input, weight, support, n_rows);

    // 2) bias broadcast into output
    const int total = n_rows * OUT_DIM;
    gcn_bias_init<<<(total + 255) / 256, 256, 0, stream>>>(bias, out, total);

    // 3) edge scatter-add
    const long long sthreads = (long long)n_edges * 16;
    const int sblocks = (int)((sthreads + 255) / 256);
    gcn_scatter<<<sblocks, 256, 0, stream>>>(support, edge_val, edge_src, edge_dst,
                                             out, n_edges);
}
